// TriangleAttention_12103217840567
// MI455X (gfx1250) — compile-verified, matrix operand order checked by reading, not run on hardware
//
#include <hip/hip_runtime.h>
#include <hip/hip_bf16.h>

typedef __bf16 bf16_t;
typedef __attribute__((ext_vector_type(16))) __bf16 v16bf;
typedef __attribute__((ext_vector_type(8)))  __bf16 v8bf;
typedef __attribute__((ext_vector_type(4)))  __bf16 v4bf;
typedef __attribute__((ext_vector_type(8)))  float  v8f;

#define NSEQ 384
#define CZ   128
#define NH   4
#define HD   32
#define MROWS (NSEQ*NSEQ)          // 147456

// ---- WMMA helpers -------------------------------------------------------
__device__ __forceinline__ v8f wmma_bf16(v16bf a, v16bf b, v8f c) {
  return __builtin_amdgcn_wmma_f32_16x16x32_bf16(
      /*neg_a=*/false, a, /*neg_b=*/false, b,
      /*c_mod=*/(short)0, c, /*reuse_a=*/false, /*reuse_b=*/false);
}

// Load a 16x32 bf16 operand row for one lane (A or B, K-contiguous storage).
// Per ISA 16-bit A layout: lanes 0-15 hold K=[0..7]+[16..23], lanes 16-31
// hold K=[8..15]+[24..31].  `row` points at the lane's row, `base` = 0 or 8.
__device__ __forceinline__ v16bf load_op_bf16(const bf16_t* row, int base) {
  v8bf lo = *(const v8bf*)(row + base);
  v8bf hi = *(const v8bf*)(row + base + 16);
  return __builtin_shufflevector(lo, hi, 0,1,2,3,4,5,6,7,8,9,10,11,12,13,14,15);
}

// Same, but source is f32 (softmax probabilities in LDS) -> convert to bf16.
// float4 loads keep the LDS traffic as ds_load_b128.
__device__ __forceinline__ v16bf load_op_f32(const float* row, int base) {
  float4 a0 = *(const float4*)(row + base);
  float4 a1 = *(const float4*)(row + base + 4);
  float4 b0 = *(const float4*)(row + base + 16);
  float4 b1 = *(const float4*)(row + base + 20);
  v16bf a;
  a[0]  = (__bf16)a0.x; a[1]  = (__bf16)a0.y; a[2]  = (__bf16)a0.z; a[3]  = (__bf16)a0.w;
  a[4]  = (__bf16)a1.x; a[5]  = (__bf16)a1.y; a[6]  = (__bf16)a1.z; a[7]  = (__bf16)a1.w;
  a[8]  = (__bf16)b0.x; a[9]  = (__bf16)b0.y; a[10] = (__bf16)b0.z; a[11] = (__bf16)b0.w;
  a[12] = (__bf16)b1.x; a[13] = (__bf16)b1.y; a[14] = (__bf16)b1.z; a[15] = (__bf16)b1.w;
  return a;
}

// ---- kernel 1: weight convert / transpose to bf16 -----------------------
// Wt[cN][k]  : cN in [0,512): cols 0..383 = qkv_w, 384..511 = gate_w
// OWt[co][k] : out_w transposed
__global__ void k_wconv(const float* __restrict__ qkv_w,
                        const float* __restrict__ gate_w,
                        const float* __restrict__ out_w,
                        bf16_t* __restrict__ Wt, bf16_t* __restrict__ OWt) {
  int idx = blockIdx.x * 256 + threadIdx.x;      // grid covers 512*128+128*128
  if (idx < 512 * 128) {
    int cN = idx >> 7, k = idx & 127;
    float v = (cN < 384) ? qkv_w[k * 384 + cN] : gate_w[k * 128 + (cN - 384)];
    Wt[idx] = (__bf16)v;
  } else {
    int i2 = idx - 512 * 128;
    int co = i2 >> 7, k = i2 & 127;
    OWt[i2] = (__bf16)out_w[k * 128 + co];
  }
}

// ---- kernel 2: LayerNorm, one 128-wide row per wave32 -------------------
__global__ void __launch_bounds__(256) k_ln(const float* __restrict__ z,
                                            const float* __restrict__ w,
                                            const float* __restrict__ b,
                                            bf16_t* __restrict__ zn) {
  int warp = threadIdx.x >> 5, lane = threadIdx.x & 31;
  int r = blockIdx.x * 8 + warp;                 // 147456 rows
  const float4 x4 = *(const float4*)(z + (size_t)r * 128 + lane * 4);
  float s = x4.x + x4.y + x4.z + x4.w;
  #pragma unroll
  for (int m = 16; m; m >>= 1) s += __shfl_xor(s, m, 32);
  float mu = s * (1.0f / 128.0f);
  float d0 = x4.x - mu, d1 = x4.y - mu, d2 = x4.z - mu, d3 = x4.w - mu;
  float v = d0 * d0 + d1 * d1 + d2 * d2 + d3 * d3;
  #pragma unroll
  for (int m = 16; m; m >>= 1) v += __shfl_xor(v, m, 32);
  float inv = rsqrtf(v * (1.0f / 128.0f) + 1e-5f);
  const float4 w4 = *(const float4*)(w + lane * 4);
  const float4 b4 = *(const float4*)(b + lane * 4);
  v4bf y;
  y[0] = (__bf16)(d0 * inv * w4.x + b4.x);
  y[1] = (__bf16)(d1 * inv * w4.y + b4.y);
  y[2] = (__bf16)(d2 * inv * w4.z + b4.z);
  y[3] = (__bf16)(d3 * inv * w4.w + b4.w);
  *(v4bf*)(zn + (size_t)r * 128 + lane * 4) = y;
}

// ---- kernel 3: fused QKV + gate GEMM  [147456x128]@[128x512] -----------
// Epilogue scatters: Q,K -> [n][h][i][d] ; V -> [n][h][d][j] ; gate sigmoid.
__global__ void __launch_bounds__(256) k_qkvg(const bf16_t* __restrict__ zn,
                                              const bf16_t* __restrict__ Wt,
                                              const float* __restrict__ qkv_b,
                                              const float* __restrict__ gate_b,
                                              bf16_t* __restrict__ Q,
                                              bf16_t* __restrict__ K,
                                              bf16_t* __restrict__ Vt,
                                              bf16_t* __restrict__ gate) {
  int lane = threadIdx.x & 31, warp = threadIdx.x >> 5;
  int w  = blockIdx.x * 8 + warp;
  int mt = w >> 5, nt = w & 31;                  // 9216 M-tiles x 32 N-tiles
  int base = (lane < 16) ? 0 : 8;
  const bf16_t* ap = zn + (size_t)(mt * 16 + (lane & 15)) * 128;
  const bf16_t* bp = Wt + (size_t)(nt * 16 + (lane & 15)) * 128;
  v8f c = {};
  #pragma unroll
  for (int kk = 0; kk < 128; kk += 32)
    c = wmma_bf16(load_op_bf16(ap + kk, base), load_op_bf16(bp + kk, base), c);

  int col   = nt * 16 + (lane & 15);
  int rbase = mt * 16 + ((lane >> 4) << 3);
  if (col < 384) {
    int which = col >> 7;                        // 0=q 1=k 2=v
    int h = (col >> 5) & 3;
    int d = col & 31;
    float bias = qkv_b[col];
    #pragma unroll
    for (int r = 0; r < 8; ++r) {
      int m = rbase + r;
      int i = m / 384, n = m - i * 384;          // m = i*384 + n
      float y = c[r] + bias;
      if (which == 0)      Q [(size_t)((n * 4 + h) * 384 + i) * 32 + d] = (__bf16)y;
      else if (which == 1) K [(size_t)((n * 4 + h) * 384 + i) * 32 + d] = (__bf16)y;
      else                 Vt[(size_t)((n * 4 + h) * 32 + d) * 384 + i] = (__bf16)y;
    }
  } else {
    int g = col - 384;
    float bias = gate_b[g];
    #pragma unroll
    for (int r = 0; r < 8; ++r) {
      float e = __expf(-(c[r] + bias));
      gate[(size_t)(rbase + r) * 128 + g] = (__bf16)__builtin_amdgcn_rcpf(1.0f + e);
    }
  }
}

// ---- kernel 4: attention, one workgroup per (n,h) -----------------------
// S kept full-i (384) x j-block (128) in LDS; softmax over i (axis=1 in the
// reference); 1/colsum folded into V columns; O accumulated in registers.
#define SPITCH 132                               // 128 + 4 pad (bank spread)
#define ATTN_SMEM (3 * 384 * 32 * 2 + 384 * SPITCH * 4)   // 276480 B < 320KB

__global__ void __launch_bounds__(256, 1) k_attn(const bf16_t* __restrict__ Qg,
                                                 const bf16_t* __restrict__ Kg,
                                                 const bf16_t* __restrict__ Vg,
                                                 bf16_t* __restrict__ Og) {
  extern __shared__ char smem[];
  bf16_t* sQ = (bf16_t*)smem;                    // 384x32
  bf16_t* sK = sQ + 384 * 32;                    // 384x32
  bf16_t* sV = sK + 384 * 32;                    // 32x384 (V transposed)
  float*  sS = (float*)(sV + 384 * 32);          // 384xSPITCH

  int nh = blockIdx.x;
  int n = nh >> 2, h = nh & 3;
  const size_t off = (size_t)nh * (384 * 32);
  // stage Q, K, Vt (24KB each, contiguous per (n,h)) into LDS
  for (int t = threadIdx.x; t < 1536; t += 256) {
    ((uint4*)sQ)[t] = ((const uint4*)(Qg + off))[t];
    ((uint4*)sK)[t] = ((const uint4*)(Kg + off))[t];
    ((uint4*)sV)[t] = ((const uint4*)(Vg + off))[t];
  }
  __syncthreads();

  int lane = threadIdx.x & 31, warp = threadIdx.x >> 5;
  int base = (lane < 16) ? 0 : 8;
  const float scale = 0.17677669529663689f;      // 1/sqrt(32)
  v8f acc[6];
  #pragma unroll
  for (int u = 0; u < 6; ++u) acc[u] = (v8f){};

  for (int jb = 0; jb < 3; ++jb) {
    int j0 = jb * 128;
    // ---- S = scale * Q K^T : 24 i-tiles x 8 j-tiles, K=32 -> 1 WMMA each
    for (int t = warp; t < 192; t += 8) {
      int it = t >> 3, jt = t & 7;
      const bf16_t* ap = sQ + (it * 16 + (lane & 15)) * 32;
      const bf16_t* bp = sK + (j0 + jt * 16 + (lane & 15)) * 32;
      v8f c = {};
      c = wmma_bf16(load_op_bf16(ap, base), load_op_bf16(bp, base), c);
      int col = jt * 16 + (lane & 15);
      int rb  = it * 16 + ((lane >> 4) << 3);
      #pragma unroll
      for (int r = 0; r < 8; ++r) sS[(rb + r) * SPITCH + col] = c[r] * scale;
    }
    __syncthreads();
    // ---- column softmax over i (384 rows); fold 1/sum into V column j
    if (threadIdx.x < 128) {
      int jl = threadIdx.x;
      float mx = -3.4e38f;
      for (int i = 0; i < 384; ++i) mx = fmaxf(mx, sS[i * SPITCH + jl]);
      float sum = 0.0f;
      for (int i = 0; i < 384; ++i) {
        float e = __expf(sS[i * SPITCH + jl] - mx);
        sS[i * SPITCH + jl] = e;
        sum += e;
      }
      float inv = __builtin_amdgcn_rcpf(sum);
      int j = j0 + jl;
      #pragma unroll
      for (int d = 0; d < 32; ++d)
        sV[d * 384 + j] = (__bf16)((float)sV[d * 384 + j] * inv);
    }
    __syncthreads();
    // ---- O += P * V : wave owns 6 of the 48 (i-tile, d-tile) outputs
    #pragma unroll
    for (int u = 0; u < 6; ++u) {
      int tile = warp + u * 8;
      int it = tile >> 1, dt = tile & 1;
      const float*  ap = sS + (it * 16 + (lane & 15)) * SPITCH;
      const bf16_t* bp = sV + (dt * 16 + (lane & 15)) * 384 + j0;
      #pragma unroll
      for (int kk = 0; kk < 128; kk += 32)
        acc[u] = wmma_bf16(load_op_f32(ap + kk, base),
                           load_op_bf16(bp + kk, base), acc[u]);
    }
    __syncthreads();
  }
  // ---- write O as bf16 rows [(i*384+n)*128 + h*32 + d]
  #pragma unroll
  for (int u = 0; u < 6; ++u) {
    int tile = warp + u * 8;
    int it = tile >> 1, dt = tile & 1;
    int col = dt * 16 + (lane & 15);
    int rb  = it * 16 + ((lane >> 4) << 3);
    int ch  = h * 32 + col;
    #pragma unroll
    for (int r = 0; r < 8; ++r) {
      int i = rb + r;
      Og[((size_t)i * 384 + n) * 128 + ch] = (__bf16)acc[u][r];
    }
  }
}

// ---- kernel 5: out-proj GEMM + bias + gate + residual -------------------
__global__ void __launch_bounds__(256) k_out(const bf16_t* __restrict__ O,
                                             const bf16_t* __restrict__ OWt,
                                             const float* __restrict__ out_b,
                                             const bf16_t* __restrict__ gate,
                                             const float* __restrict__ z,
                                             float* __restrict__ out) {
  int lane = threadIdx.x & 31, warp = threadIdx.x >> 5;
  int w  = blockIdx.x * 8 + warp;
  int mt = w >> 3, nt = w & 7;                   // 9216 M-tiles x 8 N-tiles
  int base = (lane < 16) ? 0 : 8;
  const bf16_t* ap = O   + (size_t)(mt * 16 + (lane & 15)) * 128;
  const bf16_t* bp = OWt + (size_t)(nt * 16 + (lane & 15)) * 128;
  v8f c = {};
  #pragma unroll
  for (int kk = 0; kk < 128; kk += 32)
    c = wmma_bf16(load_op_bf16(ap + kk, base), load_op_bf16(bp + kk, base), c);
  int col = nt * 16 + (lane & 15);
  int rb  = mt * 16 + ((lane >> 4) << 3);
  float bias = out_b[col];
  #pragma unroll
  for (int r = 0; r < 8; ++r) {
    size_t idx = (size_t)(rb + r) * 128 + col;
    out[idx] = z[idx] + (float)gate[idx] * (c[r] + bias);
  }
}

// ---- launch -------------------------------------------------------------
extern "C" void kernel_launch(void* const* d_in, const int* in_sizes, int n_in,
                              void* d_out, int out_size, void* d_ws, size_t ws_size,
                              hipStream_t stream) {
  const float* z      = (const float*)d_in[0];
  const float* ln_w   = (const float*)d_in[1];
  const float* ln_b   = (const float*)d_in[2];
  const float* qkv_w  = (const float*)d_in[3];
  const float* qkv_b  = (const float*)d_in[4];
  const float* out_w  = (const float*)d_in[5];
  const float* out_b  = (const float*)d_in[6];
  const float* gate_w = (const float*)d_in[7];
  const float* gate_b = (const float*)d_in[8];
  float* out = (float*)d_out;

  bf16_t* ws = (bf16_t*)d_ws;
  const size_t NE = (size_t)MROWS * 128;         // 18,874,368 elements
  bf16_t* zn   = ws;
  bf16_t* Q    = zn   + NE;
  bf16_t* K    = Q    + NE;
  bf16_t* Vt   = K    + NE;
  bf16_t* gate = Vt   + NE;
  bf16_t* O    = gate + NE;
  bf16_t* Wt   = O    + NE;                      // 512*128
  bf16_t* OWt  = Wt   + 512 * 128;               // 128*128

  k_wconv<<<(512 * 128 + 128 * 128) / 256, 256, 0, stream>>>(qkv_w, gate_w, out_w, Wt, OWt);
  k_ln   <<<MROWS / 8, 256, 0, stream>>>(z, ln_w, ln_b, zn);
  k_qkvg <<<(MROWS / 16) * (512 / 16) / 8, 256, 0, stream>>>(zn, Wt, qkv_b, gate_b, Q, K, Vt, gate);
  k_attn <<<NSEQ * NH, 256, ATTN_SMEM, stream>>>(Q, K, Vt, O);
  k_out  <<<(MROWS / 16) * (128 / 16) / 8, 256, 0, stream>>>(O, OWt, out_b, gate, z, out);
}
